// PathAttention_66219805769894
// MI455X (gfx1250) — matrix operand order repaired and checked by reading, not compiled, not run on hardware
//
#include <hip/hip_runtime.h>
#include <hip/hip_bf16.h>

typedef __attribute__((ext_vector_type(16))) __bf16 v16bf;
typedef __attribute__((ext_vector_type(8)))  float  v8f;

// ---------- constants for this problem ----------
#define BB   128
#define NN1  196
#define NN2  64
#define DD1  2048
#define DD2  1024
#define HH   128           // per-head dim
#define NBH  (BB*8)        // 1024

__device__ __forceinline__ unsigned short f2bfu(float x) {
  unsigned u = __float_as_uint(x);
  unsigned r = u + 0x7FFFu + ((u >> 16) & 1u);
  return (unsigned short)(r >> 16);
}

// A fragment: lane L holds row (L&15); lanes0-15: K={0..7,16..23}, lanes16-31: +8
__device__ __forceinline__ v16bf load_frag_a(const unsigned short* p, int half) {
  union { v16bf v; unsigned u[8]; } f;
  const unsigned* q = reinterpret_cast<const unsigned*>(p);
  int kb2 = half * 4;
  f.u[0] = q[kb2 + 0]; f.u[1] = q[kb2 + 1]; f.u[2] = q[kb2 + 2]; f.u[3] = q[kb2 + 3];
  f.u[4] = q[kb2 + 8]; f.u[5] = q[kb2 + 9]; f.u[6] = q[kb2 + 10]; f.u[7] = q[kb2 + 11];
  return f.v;
}
// B fragment (stored [n][k] in LDS): lane L holds col (L&15); lanes0-15: K={0..7,16..23}, lanes16-31: +8
__device__ __forceinline__ v16bf load_frag_b(const unsigned short* p, int half) {
  union { v16bf v; unsigned u[8]; } f;
  const unsigned* q = reinterpret_cast<const unsigned*>(p);
  int kb2 = half * 4;
#pragma unroll
  for (int i = 0; i < 8; ++i) f.u[i] = q[kb2 + (i < 4 ? i : i + 4)];
  return f.v;
}

__device__ __forceinline__ v8f wmma_bf16(v16bf a, v16bf b, v8f c) {
  return __builtin_amdgcn_wmma_f32_16x16x32_bf16(false, a, false, b, (short)0, c, false, false);
}

// ---------------------------------------------------------------------------
// Tiled GEMM: out = A[M,K](f32) @ B[K,N](f32) + bias[N], bf16 WMMA f32-accum.
// MODE 0: out = C+bias.  MODE 1: out = mul * (C+bias)  (fused emb1 epilogue).
// WG: 256 thr = 8 waves; WG tile 128x128; wave tile 32x64 (2x4 WMMA frags).
// __launch_bounds__(256,1): allow enough VGPRs to hold all 8 accumulator
// fragments (64 VGPRs) + A/B fragments without scratch spills.
// M % 128 == 0, N % 128 == 0, K % 32 == 0.
// ---------------------------------------------------------------------------
template <int MODE>
__global__ void __launch_bounds__(256, 1)
gemm_bf16_kernel(const float* __restrict__ A, int lda,
                 const float* __restrict__ B, int ldb,
                 const float* __restrict__ bias,
                 float* __restrict__ out,
                 const float* __restrict__ mul,
                 int M, int N, int K) {
  __shared__ __align__(16) unsigned short lA[128 * 32];  // [row][k]
  __shared__ __align__(16) unsigned short lB[128 * 32];  // [n][k] (transposed)

  const int tid  = threadIdx.x;
  const int lane = tid & 31;
  const int half = lane >> 4;
  const int l15  = lane & 15;
  const int wave = tid >> 5;
  const int wr   = wave >> 1;   // 0..3  -> row offset wr*32
  const int wc   = wave & 1;    // 0..1  -> col offset wc*64
  const int rowBase = blockIdx.y * 128;
  const int colBase = blockIdx.x * 128;

  v8f acc[2][4];
#pragma unroll
  for (int a = 0; a < 2; ++a)
#pragma unroll
    for (int b = 0; b < 4; ++b)
#pragma unroll
      for (int r = 0; r < 8; ++r) acc[a][b][r] = 0.0f;

  const int kt_n = K >> 5;
  for (int kt = 0; kt < kt_n; ++kt) {
    // --- stage A tile 128x32 (f32 -> bf16) ---
    {
      int r0 = tid >> 3;            // 0..31
      int k4 = (tid & 7) << 2;      // 0..28 step 4
#pragma unroll
      for (int rr = 0; rr < 4; ++rr) {
        int r = r0 + rr * 32;
        const float4 f = *reinterpret_cast<const float4*>(
            A + (size_t)(rowBase + r) * lda + kt * 32 + k4);
        unsigned short* d = lA + r * 32 + k4;
        d[0] = f2bfu(f.x); d[1] = f2bfu(f.y); d[2] = f2bfu(f.z); d[3] = f2bfu(f.w);
      }
    }
    // --- stage B tile 32x128 transposed -> [n][k] ---
    {
      int k0 = tid >> 5;            // 0..7
      int n4 = (tid & 31) << 2;     // 0..124 step 4
#pragma unroll
      for (int kk = 0; kk < 4; ++kk) {
        int k = k0 + kk * 8;
        const float4 f = *reinterpret_cast<const float4*>(
            B + (size_t)(kt * 32 + k) * ldb + colBase + n4);
        lB[(n4 + 0) * 32 + k] = f2bfu(f.x);
        lB[(n4 + 1) * 32 + k] = f2bfu(f.y);
        lB[(n4 + 2) * 32 + k] = f2bfu(f.z);
        lB[(n4 + 3) * 32 + k] = f2bfu(f.w);
      }
    }
    __syncthreads();

    v16bf aF[2], bF[4];
#pragma unroll
    for (int fi = 0; fi < 2; ++fi)
      aF[fi] = load_frag_a(lA + (wr * 32 + fi * 16 + l15) * 32, half);
#pragma unroll
    for (int fj = 0; fj < 4; ++fj)
      bF[fj] = load_frag_b(lB + (wc * 64 + fj * 16 + l15) * 32, half);

#pragma unroll
    for (int fi = 0; fi < 2; ++fi)
#pragma unroll
      for (int fj = 0; fj < 4; ++fj)
        acc[fi][fj] = wmma_bf16(aF[fi], bF[fj], acc[fi][fj]);
    __syncthreads();
  }

  // --- epilogue: C/D frag: element r -> row = half*8 + r, col = l15 ---
#pragma unroll
  for (int fi = 0; fi < 2; ++fi) {
#pragma unroll
    for (int fj = 0; fj < 4; ++fj) {
      int col = colBase + wc * 64 + fj * 16 + l15;
      float bv = bias[col];
#pragma unroll
      for (int r = 0; r < 8; ++r) {
        int row = rowBase + wr * 32 + fi * 16 + half * 8 + r;
        size_t o = (size_t)row * N + col;
        float v = acc[fi][fj][r] + bv;
        if (MODE == 0) out[o] = v;
        else           out[o] = mul[o] * v;
      }
    }
  }
}

// ---------------------------------------------------------------------------
// Generic mid-axis sum:  dst[o,in] = scale * sum_r src[o,r,in]
// ---------------------------------------------------------------------------
__global__ void midsum_kernel(const float* __restrict__ src, float* __restrict__ dst,
                              int outer, int len, int inner, float scale) {
  int g = blockIdx.x * blockDim.x + threadIdx.x;
  if (g >= outer * inner) return;
  int o = g / inner, in = g - o * inner;
  const float* p = src + (size_t)o * len * inner + in;
  float s = 0.0f;
  for (int r = 0; r < len; ++r) s += p[(size_t)r * inner];
  dst[g] = s * scale;
}

// ---------------------------------------------------------------------------
// aff[b,h,i,j] = (T1[j,i] * T2[i,j] * cw[h] + cb[h]) / sqrt(H)
//   T1[j,i] = sum_m a1p[b,h,j,m] * tanh(a2psum[m]*a1sum[i])
//   T2[i,j] = sum_n a2p[b,h,i,n] * tanh(a1psum[n]*a2sum[j])
// One WG per (b,h).
// ---------------------------------------------------------------------------
__global__ void aff_kernel(const float* __restrict__ a1p, const float* __restrict__ a2p,
                           const float* __restrict__ a1sum, const float* __restrict__ a2sum,
                           const float* __restrict__ a1psum, const float* __restrict__ a2psum,
                           const float* __restrict__ cw, const float* __restrict__ cb,
                           float* __restrict__ aff) {
  const int bh = blockIdx.x;
  const int h  = bh & 7;
  const int tid = threadIdx.x;

  __shared__ float s_a1sum[NN1], s_a2psum[NN1];
  __shared__ float s_a1psum[NN2], s_a2sum[NN2];
  __shared__ float L2s[NN2 * NN2];     // 16 KB
  __shared__ float L1t[NN1 * 16];      // 12.25 KB
  __shared__ float T2t[16 * NN2];      // 4 KB

  for (int i = tid; i < NN1; i += 256) {
    s_a1sum[i]  = a1sum[bh * NN1 + i];
    s_a2psum[i] = a2psum[bh * NN1 + i];
  }
  for (int i = tid; i < NN2; i += 256) {
    s_a1psum[i] = a1psum[bh * NN2 + i];
    s_a2sum[i]  = a2sum[bh * NN2 + i];
  }
  __syncthreads();
  for (int idx = tid; idx < NN2 * NN2; idx += 256) {
    int n = idx >> 6, j = idx & 63;
    L2s[idx] = tanhf(s_a1psum[n] * s_a2sum[j]);
  }
  __syncthreads();

  const float cwh = cw[h], cbh = cb[h];
  const float inv_sqrt_h = 0.08838834764831845f;  // 1/sqrt(128)
  const float* a1pB = a1p + (size_t)bh * NN2 * NN1;  // [j][m]
  const float* a2pB = a2p + (size_t)bh * NN1 * NN2;  // [i][n]
  float* affB = aff + (size_t)bh * NN1 * NN2;

  for (int it = 0; it < 13; ++it) {
    int i0 = it * 16;
    int ni = NN1 - i0; if (ni > 16) ni = 16;
    for (int idx = tid; idx < NN1 * 16; idx += 256) {
      int m = idx >> 4, ii = idx & 15;
      L1t[idx] = (ii < ni) ? tanhf(s_a2psum[m] * s_a1sum[i0 + ii]) : 0.0f;
    }
    for (int idx = tid; idx < 16 * NN2; idx += 256) {
      int ii = idx >> 6, j = idx & 63;
      float s = 0.0f;
      if (ii < ni) {
        const float* row = a2pB + (size_t)(i0 + ii) * NN2;
        for (int n = 0; n < NN2; ++n) s += row[n] * L2s[n * NN2 + j];
      }
      T2t[idx] = s;
    }
    __syncthreads();
    for (int idx = tid; idx < 16 * NN2; idx += 256) {
      int ii = idx >> 6, j = idx & 63;
      if (ii < ni) {
        const float* row = a1pB + (size_t)j * NN1;
        float t1 = 0.0f;
        for (int m = 0; m < NN1; ++m) t1 += row[m] * L1t[m * 16 + ii];
        affB[(i0 + ii) * NN2 + j] = (t1 * T2t[idx] * cwh + cbh) * inv_sqrt_h;
      }
    }
    __syncthreads();
  }
}

// softmax over N1 (rows masked by v_mask[b,i]); one WG of 64 threads per (b,h)
__global__ void softmax_n1_kernel(const float* __restrict__ aff,
                                  const unsigned char* __restrict__ vmask,
                                  float* __restrict__ satt1) {
  const int bh = blockIdx.x, b = bh >> 3;
  const int j = threadIdx.x;  // 0..63
  __shared__ float flag[NN1];
  for (int i = j; i < NN1; i += 64) flag[i] = vmask[b * NN1 + i] ? 1.0f : 0.0f;
  __syncthreads();
  const float* base = aff + (size_t)bh * NN1 * NN2;
  float* ob = satt1 + (size_t)bh * NN1 * NN2;
  float m = -3.4e38f;
  for (int i = 0; i < NN1; ++i) {
    float v = (flag[i] != 0.0f) ? -1e9f : base[i * NN2 + j];
    m = fmaxf(m, v);
  }
  float s = 0.0f;
  for (int i = 0; i < NN1; ++i) {
    float v = (flag[i] != 0.0f) ? -1e9f : base[i * NN2 + j];
    s += __expf(v - m);
  }
  float inv = 1.0f / s;
  for (int i = 0; i < NN1; ++i) {
    float v = (flag[i] != 0.0f) ? -1e9f : base[i * NN2 + j];
    ob[i * NN2 + j] = __expf(v - m) * inv;
  }
}

// softmax over N2 (cols masked by q_mask[b,j]), written transposed -> satt2[b,h,j,i]
__global__ void softmax_n2_kernel(const float* __restrict__ aff,
                                  const unsigned char* __restrict__ qmask,
                                  float* __restrict__ satt2) {
  const int bh = blockIdx.x, b = bh >> 3;
  const int tid = threadIdx.x;
  __shared__ float flag[NN2];
  if (tid < NN2) flag[tid] = qmask[b * NN2 + tid] ? 1.0f : 0.0f;
  __syncthreads();
  for (int i = tid; i < NN1; i += 256) {
    const float* row = aff + (size_t)bh * NN1 * NN2 + i * NN2;
    float* ob = satt2 + (size_t)bh * NN2 * NN1;
    float m = -3.4e38f;
    for (int jj = 0; jj < NN2; ++jj) {
      float v = (flag[jj] != 0.0f) ? -1e9f : row[jj];
      m = fmaxf(m, v);
    }
    float s = 0.0f;
    for (int jj = 0; jj < NN2; ++jj) {
      float v = (flag[jj] != 0.0f) ? -1e9f : row[jj];
      s += __expf(v - m);
    }
    float inv = 1.0f / s;
    for (int jj = 0; jj < NN2; ++jj) {
      float v = (flag[jj] != 0.0f) ? -1e9f : row[jj];
      ob[jj * NN1 + i] = __expf(v - m) * inv;
    }
  }
}

// m1[b,i,h,c] = sum_n satt1[b,h,i,n] * x2[b,n,h*128+c] * gate1[b,h*128+c]
__global__ void m1_kernel(const float* __restrict__ satt1, const float* __restrict__ x2,
                          const float* __restrict__ gate1, float* __restrict__ m1) {
  const int bh = blockIdx.x, b = bh >> 3, h = bh & 7;
  const int tid = threadIdx.x;
  __shared__ float Dg[NN2 * HH];  // 32 KB
  for (int idx = tid; idx < NN2 * HH; idx += 256) {
    int n = idx >> 7, c = idx & 127;
    Dg[idx] = x2[((size_t)(b * NN2 + n)) * DD2 + h * HH + c] *
              gate1[b * DD2 + h * HH + c];
  }
  __syncthreads();
  const float* sb = satt1 + (size_t)bh * NN1 * NN2;
  for (int idx = tid; idx < NN1 * HH; idx += 256) {
    int i = idx >> 7, c = idx & 127;
    const float* srow = sb + i * NN2;
    float s = 0.0f;
    for (int n = 0; n < NN2; ++n) s += srow[n] * Dg[n * HH + c];
    m1[((size_t)(b * NN1 + i) * 8 + h) * HH + c] = s;
  }
}

// emb2[b,n,h*128+c] = x2[..] * gate2[..] * sum_i satt2[b,h,n,i] * Qf[b,i,h*128+c]
__global__ void m2_emb2_kernel(const float* __restrict__ satt2, const float* __restrict__ Qf,
                               const float* __restrict__ gate2, const float* __restrict__ x2,
                               float* __restrict__ emb2) {
  const int bh = blockIdx.x, b = bh >> 3, h = bh & 7;
  const int tid = threadIdx.x;
  const float* sb = satt2 + (size_t)bh * NN2 * NN1;
  for (int idx = tid; idx < NN2 * HH; idx += 256) {
    int j = idx >> 7, c = idx & 127;
    float g = gate2[b * DD2 + h * HH + c];
    const float* srow = sb + j * NN1;
    const float* qcol = Qf + (size_t)b * NN1 * DD2 + h * HH + c;
    float s = 0.0f;
    for (int i = 0; i < NN1; ++i) s += srow[i] * qcol[(size_t)i * DD2];
    size_t o = ((size_t)(b * NN2 + j)) * DD2 + h * HH + c;
    emb2[o] = x2[o] * (s * g);
  }
}

// ---------------------------------------------------------------------------
extern "C" void kernel_launch(void* const* d_in, const int* in_sizes, int n_in,
                              void* d_out, int out_size, void* d_ws, size_t ws_size,
                              hipStream_t stream) {
  const float* x1   = (const float*)d_in[0];
  const float* x2   = (const float*)d_in[1];
  const float* a1   = (const float*)d_in[2];
  const float* a2   = (const float*)d_in[3];
  const float* a1p  = (const float*)d_in[4];
  const float* a2p  = (const float*)d_in[5];
  const unsigned char* vmask = (const unsigned char*)d_in[6];
  const unsigned char* qmask = (const unsigned char*)d_in[7];
  const float* Wq = (const float*)d_in[8];
  const float* bq = (const float*)d_in[9];
  const float* Wr = (const float*)d_in[10];
  const float* br = (const float*)d_in[11];
  const float* cw = (const float*)d_in[12];
  const float* cb = (const float*)d_in[13];

  // output layout: emb1 | emb2 | satt1 | satt2
  float* emb1  = (float*)d_out;
  float* emb2  = emb1 + (size_t)BB * NN1 * DD1;       // 51,380,224
  float* satt1 = emb2 + (size_t)BB * NN2 * DD2;       //  8,388,608
  float* satt2 = satt1 + (size_t)NBH * NN1 * NN2;     // 12,845,056

  // workspace layout (floats)
  float* ws      = (float*)d_ws;
  float* Qf      = ws;                                    // 25,690,112
  float* m1      = Qf + (size_t)BB * NN1 * DD2;           // 25,690,112
  float* aff     = m1 + (size_t)BB * NN1 * DD2;           // 12,845,056
  float* gate1   = aff + (size_t)NBH * NN1 * NN2;         //    131,072
  float* gate2   = gate1 + (size_t)BB * DD2;              //    131,072
  float* a1sumW  = gate2 + (size_t)BB * DD2;              //    200,704
  float* a2sumW  = a1sumW + (size_t)NBH * NN1;            //     65,536
  float* a1psumW = a2sumW + (size_t)NBH * NN2;            //     65,536
  float* a2psumW = a1psumW + (size_t)NBH * NN2;           //    200,704

  auto grid1d = [](int total) { return dim3((total + 255) / 256); };

  // --- small reductions ---
  midsum_kernel<<<grid1d(NBH * NN1), 256, 0, stream>>>(a1, a1sumW, NBH, NN2, NN1, 1.0f);
  midsum_kernel<<<grid1d(NBH * NN2), 256, 0, stream>>>(a2, a2sumW, NBH, NN1, NN2, 1.0f);
  midsum_kernel<<<grid1d(NBH * NN2), 256, 0, stream>>>(a1p, a1psumW, NBH * NN2, NN1, 1, 1.0f);
  midsum_kernel<<<grid1d(NBH * NN1), 256, 0, stream>>>(a2p, a2psumW, NBH * NN1, NN2, 1, 1.0f);
  midsum_kernel<<<grid1d(BB * DD2), 256, 0, stream>>>(x2, gate2, BB, NN2, DD2, 1.0f / 64.0f);

  // --- GEMM1: Qf = x1 @ Wq + bq (bf16 WMMA) ---
  gemm_bf16_kernel<0><<<dim3(DD2 / 128, (BB * NN1) / 128), 256, 0, stream>>>(
      x1, DD1, Wq, DD2, bq, Qf, nullptr, BB * NN1, DD2, DD1);

  // gate1[b,c] = mean_i Qf[b,i,c]
  midsum_kernel<<<grid1d(BB * DD2), 256, 0, stream>>>(Qf, gate1, BB, NN1, DD2, 1.0f / 196.0f);

  // --- path affinity ---
  aff_kernel<<<NBH, 256, 0, stream>>>(a1p, a2p, a1sumW, a2sumW, a1psumW, a2psumW, cw, cb, aff);

  // --- softmaxes (write directly into d_out) ---
  softmax_n1_kernel<<<NBH, 64, 0, stream>>>(aff, vmask, satt1);
  softmax_n2_kernel<<<NBH, 256, 0, stream>>>(aff, qmask, satt2);

  // --- per-head attention applies ---
  m1_kernel<<<NBH, 256, 0, stream>>>(satt1, x2, gate1, m1);
  m2_emb2_kernel<<<NBH, 256, 0, stream>>>(satt2, Qf, gate2, x2, emb2);

  // --- GEMM2 + fused emb1 = x1 * (m1_reform @ Wr + br) ---
  gemm_bf16_kernel<1><<<dim3(DD1 / 128, (BB * NN1) / 128), 256, 0, stream>>>(
      m1, DD2, Wr, DD1, br, emb1, x1, BB * NN1, DD1, DD2);
}
